// SMCrossBaby_1_953482739801
// MI455X (gfx1250) — matrix operand order repaired and checked by reading, not compiled, not run on hardware
//
#include <hip/hip_runtime.h>
#include <hip/hip_bf16.h>

// B=16, S=128, V=32000, E=512, M=1024
#define BB 16
#define SS 128
#define VV 32000
#define EE 512
#define MM 1024

typedef __bf16 bf16_t;
typedef bf16_t v16bf __attribute__((ext_vector_type(16)));
typedef float  v8f   __attribute__((ext_vector_type(8)));

typedef unsigned int u32x4 __attribute__((ext_vector_type(4)));
typedef int          i32x4 __attribute__((ext_vector_type(4)));
typedef int          i32x8 __attribute__((ext_vector_type(8)));

#if defined(__has_builtin)
#  if __has_builtin(__builtin_amdgcn_tensor_load_to_lds) && \
      __has_builtin(__builtin_amdgcn_s_wait_tensorcnt)
#    define HAVE_TDM 1
#  else
#    define HAVE_TDM 0
#  endif
#else
#  define HAVE_TDM 0
#endif

__device__ __forceinline__ v8f wmma_bf16(v16bf a, v16bf b, v8f c) {
    return __builtin_amdgcn_wmma_f32_16x16x32_bf16(
        /*neg_a=*/false, a, /*neg_b=*/false, b,
        /*c_mod=*/(short)0, c, /*reuse_a=*/false, /*reuse_b=*/false);
}

// ---- Fragment loaders (f32 source, convert to bf16) ----
// A: 16x32 (MxK), row-major source with leading dim `ld` (floats).
__device__ __forceinline__ v16bf load_a_rm(const float* base, int ld) {
    int lane = threadIdx.x & 31;
    int half = lane >> 4;
    int m    = lane & 15;
    const float* rp = base + (size_t)m * ld + 16 * half;
    v16bf a;
#pragma unroll
    for (int v = 0; v < 8; ++v) {
        int kk = 2 * v;
        float2 f = *(const float2*)(rp + kk);
        a[2 * v]     = (bf16_t)f.x;
        a[2 * v + 1] = (bf16_t)f.y;
    }
    return a;
}

// B: 32x16 (KxN) from a weight stored [N, K] row-major (ldk = K-stride).
__device__ __forceinline__ v16bf load_b_nk(const float* W, int ldk) {
    int lane = threadIdx.x & 31;
    int n    = lane & 15;
    int koff = (lane >> 4) * 16;
    const float* p = W + (size_t)n * ldk + koff;
    v16bf b;
#pragma unroll
    for (int j = 0; j < 16; j += 4) {
        float4 f = *(const float4*)(p + j);
        b[j]     = (bf16_t)f.x;
        b[j + 1] = (bf16_t)f.y;
        b[j + 2] = (bf16_t)f.z;
        b[j + 3] = (bf16_t)f.w;
    }
    return b;
}

// B: 32x16 (KxN) from a matrix stored [K, N] row-major (ldn = N-stride).
__device__ __forceinline__ v16bf load_b_kn(const float* W, int ldn) {
    int lane = threadIdx.x & 31;
    int n    = lane & 15;
    int koff = (lane >> 4) * 16;
    const float* p = W + (size_t)koff * ldn + n;
    v16bf b;
#pragma unroll
    for (int j = 0; j < 16; ++j) b[j] = (bf16_t)p[(size_t)j * ldn];
    return b;
}

#if HAVE_TDM
// Issue a TDM load of a 16(row) x 128(col) f32 tile of x into LDS.
// D# per ISA 8.3/8.4: group0 = {count/flags, lds_addr, global_addr lo,
// global_addr hi | type=2}; group1 packs data_size=4B, tensor_dim0=VV,
// tensor_dim1=2048, tile_dim0=128, tile_dim1=16, tensor_dim0_stride=VV.
__device__ __forceinline__ void tdm_load_x_tile(const float* gsrc, void* lds_dst) {
    unsigned long long ga = (unsigned long long)(uintptr_t)gsrc;
    unsigned int lds_addr = (unsigned int)(uintptr_t)lds_dst;  // low 32 bits = LDS byte addr
    u32x4 g0;
    g0[0] = 1u;                                   // count=1 (valid user descriptor)
    g0[1] = lds_addr;                             // bits [63:32]
    g0[2] = (unsigned int)ga;                     // global_addr[31:0]
    g0[3] = (unsigned int)((ga >> 32) & 0x01FFFFFFu) | (2u << 30);  // addr[56:32], type=2
    i32x8 g1;
    g1[0] = 0x00020000;                           // data_size=2 (4 bytes)
    g1[1] = (int)((unsigned)(VV & 0xFFFF) << 16); // tensor_dim0 lo16 -> [63:48]
    g1[2] = (int)((unsigned)(2048 & 0xFFFF) << 16); // dim0 hi16=0 | tensor_dim1 lo16
    g1[3] = (int)((unsigned)128 << 16);           // dim1 hi16=0 | tile_dim0=128
    g1[4] = 16;                                   // tile_dim1=16 | tile_dim2=0
    g1[5] = VV;                                   // tensor_dim0_stride lo32
    g1[6] = 0;                                    // stride0 hi16 | stride1 lo16
    g1[7] = 0;                                    // stride1 hi32
    i32x4 z4 = {0, 0, 0, 0};
#if __clang_major__ >= 23
    i32x8 z8 = {0, 0, 0, 0, 0, 0, 0, 0};
    __builtin_amdgcn_tensor_load_to_lds(g0, g1, z4, z4, z8, 0);
#else
    __builtin_amdgcn_tensor_load_to_lds(g0, g1, z4, z4, 0);
#endif
}
#endif

// ---------------------------------------------------------------------------
// K1: h[row, e] = relu(sum_v x[row, v] * w_embed[e, v] + b_embed[e])
// rows = B*S = 2048, K = 32000. 128 blocks (16-row strips), 8 waves, each wave
// owns 64 cols. x double-buffered in LDS via TDM (wave 0 issues, all compute).
// ---------------------------------------------------------------------------
__global__ __launch_bounds__(256) void k_embed(const float* __restrict__ x,
                                               const float* __restrict__ w_embed,
                                               const float* __restrict__ b_embed,
                                               float* __restrict__ h) {
    __shared__ float xs[2][16 * 128];  // 16 KB double buffer
    int row0 = blockIdx.x * 16;
    int wv   = threadIdx.x >> 5;
    int lane = threadIdx.x & 31;
    int col0 = wv * 64;
    const float* xstrip = x + (size_t)row0 * VV;
    v8f acc[4] = {};

#if HAVE_TDM
    if (wv == 0) tdm_load_x_tile(xstrip, &xs[0][0]);
#endif
    for (int it = 0; it < VV / 128; ++it) {
        const int kb = it * 128;
        float* buf = &xs[it & 1][0];
#if HAVE_TDM
        if (wv == 0) {
            if (it + 1 < VV / 128) {
                tdm_load_x_tile(xstrip + kb + 128, &xs[(it + 1) & 1][0]);
                __builtin_amdgcn_s_wait_tensorcnt(1);  // oldest (tile `it`) done
            } else {
                __builtin_amdgcn_s_wait_tensorcnt(0);
            }
        }
        __syncthreads();  // publish LDS tile `it` to all waves
#else
        __syncthreads();
        {   // cooperative fallback: 2048 floats / 256 threads
            const float* src = xstrip + kb;
#pragma unroll
            for (int r = 0; r < 2; ++r) {
                int idx = (threadIdx.x + r * 256) * 4;
                int rr  = idx >> 7;
                int cc  = idx & 127;
                *(float4*)(&buf[idx]) = *(const float4*)(src + (size_t)rr * VV + cc);
            }
        }
        __syncthreads();
#endif
#pragma unroll
        for (int ks = 0; ks < 128; ks += 32) {
            v16bf a = load_a_rm(&buf[ks], 128);
#pragma unroll
            for (int ct = 0; ct < 4; ++ct) {
                const float* wp = w_embed + (size_t)(col0 + ct * 16) * VV + (kb + ks);
                v16bf b = load_b_nk(wp, VV);
                acc[ct] = wmma_bf16(a, b, acc[ct]);
            }
        }
        __syncthreads();  // all waves done with buf before TDM overwrites it
    }
    int half = lane >> 4, n = lane & 15;
#pragma unroll
    for (int ct = 0; ct < 4; ++ct) {
        int col  = col0 + ct * 16 + n;
        float bv = b_embed[col];
#pragma unroll
        for (int v = 0; v < 8; ++v) {
            int m     = v + 8 * half;
            float val = acc[ct][v] + bv;
            h[(size_t)(row0 + m) * EE + col] = val > 0.f ? val : 0.f;
        }
    }
}

// K2: rs[row] = sum_e h[row, e]  (one wave per row)
__global__ __launch_bounds__(256) void k_rowsum(const float* __restrict__ h,
                                                float* __restrict__ rs) {
    int w    = (blockIdx.x * 256 + threadIdx.x) >> 5;  // 0..2047
    int lane = threadIdx.x & 31;
    const float* r = h + (size_t)w * EE;
    float s = 0.f;
    for (int i = lane; i < EE; i += 32) s += r[i];
#pragma unroll
    for (int o = 16; o > 0; o >>= 1) s += __shfl_xor(s, o, 32);
    if (lane == 0) rs[w] = s;
}

// K3: A_all[b, e, j] = sum_k rs[b, k] * w_reduce[e, k*512 + j]
// For each e: (16x512) = rs(16x128) @ Wr[e](128x512). w_reduce read ONCE.
__global__ __launch_bounds__(256) void k_stageC(const float* __restrict__ rs,
                                                const float* __restrict__ w_reduce,
                                                float* __restrict__ A_all) {
    int w    = (blockIdx.x * 256 + threadIdx.x) >> 5;
    int e    = w >> 5;
    int j0   = (w & 31) * 16;
    int lane = threadIdx.x & 31;
    const float* Wr = w_reduce + (size_t)e * (SS * EE);
    v8f acc = {};
#pragma unroll
    for (int k0 = 0; k0 < SS; k0 += 32) {
        v16bf a = load_a_rm(rs + k0, SS);
        v16bf b = load_b_kn(Wr + (size_t)k0 * EE + j0, EE);
        acc = wmma_bf16(a, b, acc);
    }
    int half = lane >> 4, n = lane & 15;
#pragma unroll
    for (int v = 0; v < 8; ++v) {
        int bidx = v + 8 * half;  // batch index = tile row
        A_all[((size_t)bidx * EE + e) * EE + j0 + n] = acc[v];
    }
}

// K4: y[b, i, e] = relu(sum_j h[b, i, j] * A_all[b, e, j] + b_reduce[e])
__global__ __launch_bounds__(256) void k_stageD(const float* __restrict__ h,
                                                const float* __restrict__ A_all,
                                                const float* __restrict__ b_reduce,
                                                float* __restrict__ y) {
    int w    = (blockIdx.x * 256 + threadIdx.x) >> 5;
    int bb   = w >> 8;
    int rem  = w & 255;
    int i0   = (rem >> 5) * 16;
    int e0   = (rem & 31) * 16;
    int lane = threadIdx.x & 31;
    const float* hb = h + (size_t)bb * SS * EE;
    const float* Ab = A_all + (size_t)bb * EE * EE;
    v8f acc = {};
    for (int k0 = 0; k0 < EE; k0 += 32) {
        v16bf a = load_a_rm(hb + (size_t)i0 * EE + k0, EE);
        v16bf b = load_b_nk(Ab + (size_t)e0 * EE + k0, EE);  // A_all[b][e][j]: [N,K]
        acc = wmma_bf16(a, b, acc);
    }
    int half = lane >> 4, n = lane & 15;
    float bv = b_reduce[e0 + n];
#pragma unroll
    for (int v = 0; v < 8; ++v) {
        int m     = v + 8 * half;
        float val = acc[v] + bv;
        y[((size_t)bb * SS + i0 + m) * EE + e0 + n] = val > 0.f ? val : 0.f;
    }
}

// K5: split-K partials of y2 = y_flat(16x65536) @ w_reduce2^T(65536x512)
__global__ __launch_bounds__(256) void k_stageE(const float* __restrict__ y,
                                                const float* __restrict__ w_reduce2,
                                                float* __restrict__ part) {
    int w     = (blockIdx.x * 256 + threadIdx.x) >> 5;
    int nt    = w >> 6;
    int kc    = w & 63;
    int n0    = nt * 16;
    int kbase = kc * 1024;
    int lane  = threadIdx.x & 31;
    v8f acc = {};
    for (int k0 = 0; k0 < 1024; k0 += 32) {
        v16bf a = load_a_rm(y + kbase + k0, SS * EE);
        v16bf b = load_b_nk(w_reduce2 + (size_t)n0 * (SS * EE) + kbase + k0, SS * EE);
        acc = wmma_bf16(a, b, acc);
    }
    int half = lane >> 4, n = lane & 15;
#pragma unroll
    for (int v = 0; v < 8; ++v) {
        int m = v + 8 * half;
        part[((size_t)kc * BB + m) * EE + n0 + n] = acc[v];
    }
}

// K6: y2[idx] = relu(sum_kc part[kc][idx] + b_reduce2)   (deterministic reduce)
__global__ __launch_bounds__(256) void k_stageE_red(const float* __restrict__ part,
                                                    const float* __restrict__ b_reduce2,
                                                    float* __restrict__ y2) {
    int idx = blockIdx.x * 256 + threadIdx.x;  // 0..8191
    float s = 0.f;
    for (int kc = 0; kc < 64; ++kc) s += part[(size_t)kc * (BB * EE) + idx];
    s += b_reduce2[idx & (EE - 1)];
    y2[idx] = s > 0.f ? s : 0.f;
}

// K7: three 16x1024x512 GEMMs: logits (no act), second (relu), zint (relu)
__global__ __launch_bounds__(256) void k_stageF1(const float* __restrict__ y2,
                                                 const float* __restrict__ wf, const float* __restrict__ bf,
                                                 const float* __restrict__ wsec, const float* __restrict__ bsec,
                                                 const float* __restrict__ wi, const float* __restrict__ bi,
                                                 float* __restrict__ logits,
                                                 float* __restrict__ second,
                                                 float* __restrict__ zint) {
    int w    = (blockIdx.x * 256 + threadIdx.x) >> 5;
    int mat  = w >> 6;
    int n0   = (w & 63) * 16;
    int lane = threadIdx.x & 31;
    const float* W  = (mat == 0) ? wf : (mat == 1) ? wsec : wi;
    const float* Bv = (mat == 0) ? bf : (mat == 1) ? bsec : bi;
    float* O        = (mat == 0) ? logits : (mat == 1) ? second : zint;
    v8f acc = {};
    for (int k0 = 0; k0 < EE; k0 += 32) {
        v16bf a = load_a_rm(y2 + k0, EE);
        v16bf b = load_b_nk(W + (size_t)n0 * EE + k0, EE);
        acc = wmma_bf16(a, b, acc);
    }
    int half = lane >> 4, n = lane & 15;
    float bias = Bv[n0 + n];
#pragma unroll
    for (int v = 0; v < 8; ++v) {
        int m     = v + 8 * half;
        float val = acc[v] + bias;
        if (mat != 0) val = val > 0.f ? val : 0.f;
        O[(size_t)m * MM + n0 + n] = val;
    }
}

// K8: row softmax over 1024 (one block per row)
__global__ __launch_bounds__(256) void k_softmax(const float* __restrict__ logits,
                                                 float* __restrict__ P) {
    __shared__ float sred[8];
    int row  = blockIdx.x;
    int lane = threadIdx.x & 31, wv = threadIdx.x >> 5;
    const float* L = logits + (size_t)row * MM;
    float mx = -3.4e38f;
    for (int i = threadIdx.x; i < MM; i += 256) mx = fmaxf(mx, L[i]);
#pragma unroll
    for (int o = 16; o > 0; o >>= 1) mx = fmaxf(mx, __shfl_xor(mx, o, 32));
    if (lane == 0) sred[wv] = mx;
    __syncthreads();
    float gmx = sred[0];
#pragma unroll
    for (int i = 1; i < 8; ++i) gmx = fmaxf(gmx, sred[i]);
    __syncthreads();
    float sm = 0.f;
    for (int i = threadIdx.x; i < MM; i += 256) sm += __expf(L[i] - gmx);
#pragma unroll
    for (int o = 16; o > 0; o >>= 1) sm += __shfl_xor(sm, o, 32);
    if (lane == 0) sred[wv] = sm;
    __syncthreads();
    float tot = 0.f;
#pragma unroll
    for (int i = 0; i < 8; ++i) tot += sred[i];
    float inv = 1.0f / tot;
    for (int i = threadIdx.x; i < MM; i += 256)
        P[(size_t)row * MM + i] = __expf(L[i] - gmx) * inv;
}

// K9: first = P(16x1024) @ mem(1024x1024); z = second*first + zint
__global__ __launch_bounds__(256) void k_stageF2(const float* __restrict__ P,
                                                 const float* __restrict__ mem,
                                                 const float* __restrict__ second,
                                                 const float* __restrict__ zint,
                                                 float* __restrict__ z) {
    int w    = (blockIdx.x * 256 + threadIdx.x) >> 5;
    int n0   = w * 16;
    int lane = threadIdx.x & 31;
    v8f acc = {};
    for (int k0 = 0; k0 < MM; k0 += 32) {
        v16bf a = load_a_rm(P + k0, MM);
        v16bf b = load_b_kn(mem + (size_t)k0 * MM + n0, MM);
        acc = wmma_bf16(a, b, acc);
    }
    int half = lane >> 4, n = lane & 15;
#pragma unroll
    for (int v = 0; v < 8; ++v) {
        int m      = v + 8 * half;
        size_t idx = (size_t)m * MM + n0 + n;
        z[idx] = second[idx] * acc[v] + zint[idx];
    }
}

// K10: out[b, v] = sum_m z[b, m] * w_out[v, m] + b_out[v]
__global__ __launch_bounds__(256) void k_stageG(const float* __restrict__ z,
                                                const float* __restrict__ w_out,
                                                const float* __restrict__ b_out,
                                                float* __restrict__ out) {
    int w    = (blockIdx.x * 256 + threadIdx.x) >> 5;
    int n0   = w * 16;
    int lane = threadIdx.x & 31;
    v8f acc = {};
    for (int k0 = 0; k0 < MM; k0 += 32) {
        v16bf a = load_a_rm(z + k0, MM);
        v16bf b = load_b_nk(w_out + (size_t)n0 * MM + k0, MM);
        acc = wmma_bf16(a, b, acc);
    }
    int half = lane >> 4, n = lane & 15;
    float bv = b_out[n0 + n];
#pragma unroll
    for (int v = 0; v < 8; ++v) {
        int m = v + 8 * half;
        out[(size_t)m * VV + n0 + n] = acc[v] + bv;
    }
}

extern "C" void kernel_launch(void* const* d_in, const int* in_sizes, int n_in,
                              void* d_out, int out_size, void* d_ws, size_t ws_size,
                              hipStream_t stream) {
    const float* x         = (const float*)d_in[0];
    const float* w_embed   = (const float*)d_in[1];
    const float* b_embed   = (const float*)d_in[2];
    const float* w_reduce  = (const float*)d_in[3];
    const float* b_reduce  = (const float*)d_in[4];
    const float* w_reduce2 = (const float*)d_in[5];
    const float* b_reduce2 = (const float*)d_in[6];
    const float* mem       = (const float*)d_in[7];
    const float* w_first   = (const float*)d_in[8];
    const float* b_first   = (const float*)d_in[9];
    const float* w_second  = (const float*)d_in[10];
    const float* b_second  = (const float*)d_in[11];
    const float* w_int     = (const float*)d_in[12];
    const float* b_int     = (const float*)d_in[13];
    const float* w_out     = (const float*)d_in[14];
    const float* b_out     = (const float*)d_in[15];
    float* out = (float*)d_out;

    float* ws     = (float*)d_ws;
    float* h      = ws;                     // 2048*512     = 1048576
    float* rs     = h + 1048576;            // 2048
    float* A_all  = rs + 2048;              // 16*512*512   = 4194304
    float* y      = A_all + 4194304;        // 2048*512     = 1048576
    float* part   = y + 1048576;            // 64*16*512    = 524288
    float* y2     = part + 524288;          // 8192
    float* logits = y2 + 8192;              // 16384
    float* second = logits + 16384;         // 16384
    float* zint   = second + 16384;         // 16384
    float* P      = zint + 16384;           // 16384
    float* z      = P + 16384;              // 16384

    dim3 blk(256);
    k_embed<<<128, blk, 0, stream>>>(x, w_embed, b_embed, h);
    k_rowsum<<<256, blk, 0, stream>>>(h, rs);
    k_stageC<<<2048, blk, 0, stream>>>(rs, w_reduce, A_all);
    k_stageD<<<512, blk, 0, stream>>>(h, A_all, b_reduce, y);
    k_stageE<<<256, blk, 0, stream>>>(y, w_reduce2, part);
    k_stageE_red<<<32, blk, 0, stream>>>(part, b_reduce2, y2);
    k_stageF1<<<24, blk, 0, stream>>>(y2, w_first, b_first, w_second, b_second,
                                      w_int, b_int, logits, second, zint);
    k_softmax<<<16, blk, 0, stream>>>(logits, P);
    k_stageF2<<<8, blk, 0, stream>>>(P, mem, second, zint, z);
    k_stageG<<<250, blk, 0, stream>>>(z, w_out, b_out, out);
}
